// MHA_79362405695575
// MI455X (gfx1250) — matrix operand order repaired and checked by reading, not compiled, not run on hardware
//
#include <hip/hip_runtime.h>
#include <hip/hip_bf16.h>
#include <math.h>

typedef __bf16 bf16_t;
typedef __attribute__((ext_vector_type(16))) __bf16 v16bf;
typedef __attribute__((ext_vector_type(8)))  float  v8f;

#define DIM   2048
#define NH    32
#define NKV   8
#define HD    64
#define SEQ   2048
#define BATCH 2
#define NTOK  (BATCH * SEQ)
#define KVDIM (NKV * HD)   // 512
#define QDIM  (NH * HD)    // 2048

union FragB { v16bf v; uint4 u[2]; };

static __device__ __forceinline__ v8f wmma_bf16(const v16bf& a, const v16bf& b, v8f c) {
    return __builtin_amdgcn_wmma_f32_16x16x32_bf16(false, a, false, b, (short)0, c, false, false);
}

// ---------------------------------------------------------------------------
// fp32 -> bf16 conversion
// ---------------------------------------------------------------------------
__global__ void cvt_f32_to_bf16(const float* __restrict__ src, bf16_t* __restrict__ dst, int n) {
    int i = blockIdx.x * blockDim.x + threadIdx.x;
    if (i < n) dst[i] = (bf16_t)src[i];
}

// ---------------------------------------------------------------------------
// V transpose: v[b,s,kvh,d] -> vT[b,kvh,d,s] so attention's P@V B-fragments
// (lane = d column, elements = consecutive keys) are contiguous 32B runs.
// ---------------------------------------------------------------------------
__global__ void transpose_v(const bf16_t* __restrict__ v, bf16_t* __restrict__ vT) {
    int idx = blockIdx.x * blockDim.x + threadIdx.x;
    if (idx >= NTOK * KVDIM) return;
    int d = idx % KVDIM;       // kvh*HD + dd
    int t = idx / KVDIM;       // b*SEQ + s
    int b = t / SEQ;
    int s = t - b * SEQ;
    int kvh = d / HD;
    int dd  = d - kvh * HD;
    vT[(((size_t)b * NKV + kvh) * HD + dd) * SEQ + s] = v[idx];
}

// ---------------------------------------------------------------------------
// Tiled bf16 WMMA GEMM: C[M,N] = A[M,K] @ B[K,N] (+bias), fp32 accumulate.
// Block tile 128x64, BK=32, 256 threads = 8 waves. Wave w: N-subtile (w&3),
// M-group (w>>2)*64 with four 16-row subtiles -> 4 accumulators, 4 WMMAs
// per K step against 5 LDS fragment loads. A staged row-major, B staged
// transposed ([n][k]); all fragment gathers are contiguous 16B LDS reads.
// ---------------------------------------------------------------------------
template<bool OUT_BF16, bool HAS_BIAS>
__global__ __launch_bounds__(256) void gemm_bf16_wmma(
    const bf16_t* __restrict__ A,
    const bf16_t* __restrict__ Bw,
    const float*  __restrict__ bias,
    void* __restrict__ Cout,
    int M, int N, int K)
{
    __shared__ bf16_t As[128][40];  // rows m, cols k (stride 80B, 16B aligned)
    __shared__ bf16_t Bs[64][40];   // rows n, cols k (transposed)

    const int tid  = threadIdx.x;
    const int lane = tid & 31;
    const int wave = tid >> 5;
    const int lo   = lane & 15;
    const int hi   = lane >> 4;

    const int m0 = blockIdx.y * 128;
    const int n0 = blockIdx.x * 64;

    const int ntile  = (wave & 3) * 16;
    const int mgroup = (wave >> 2) * 64;

    v8f acc[4] = {{}, {}, {}, {}};

    for (int k0 = 0; k0 < K; k0 += 32) {
        // --- cooperative stage: A tile (128x32), 32B per thread
        {
            int r = tid >> 1;              // 0..127
            int c = (tid & 1) * 16;        // 0, 16
            *(uint4*)&As[r][c]     = *(const uint4*)(A + (size_t)(m0 + r) * K + k0 + c);
            *(uint4*)&As[r][c + 8] = *(const uint4*)(A + (size_t)(m0 + r) * K + k0 + c + 8);
        }
        // --- cooperative stage: B tile (32x64) transposed into Bs[n][k]
        {
            int kr = tid >> 3;             // 0..31
            int nc = (tid & 7) * 8;        // 0..56
            const bf16_t* src = Bw + (size_t)(k0 + kr) * N + n0 + nc;
#pragma unroll
            for (int j = 0; j < 8; ++j) Bs[nc + j][kr] = src[j];
        }
        if (k0 + 32 < K) {
            __builtin_prefetch(A  + (size_t)(m0 + (tid >> 1)) * K + k0 + 32, 0, 0);
            __builtin_prefetch(Bw + (size_t)(k0 + 32 + (tid >> 3)) * N + n0, 0, 0);
        }
        __syncthreads();

        // --- B fragment: lane lo = N col; K runs [hi*8 .. +7], [16+hi*8 .. +7]
        FragB fb;
        {
            int nrow = ntile + lo;
            fb.u[0] = *(const uint4*)&Bs[nrow][hi * 8];
            fb.u[1] = *(const uint4*)&Bs[nrow][16 + hi * 8];
        }
        // --- A fragments + WMMA per M-subtile
#pragma unroll
        for (int s = 0; s < 4; ++s) {
            FragB fa;
            int row = mgroup + s * 16 + lo;
            fa.u[0] = *(const uint4*)&As[row][hi * 8];
            fa.u[1] = *(const uint4*)&As[row][16 + hi * 8];
            acc[s] = wmma_bf16(fa.v, fb.v, acc[s]);
        }
        __syncthreads();
    }

    // --- epilogue: C layout M = r + 8*hi, N = lane lo
    const int col = n0 + ntile + lo;
    float bval = 0.0f;
    if constexpr (HAS_BIAS) bval = bias[col];
#pragma unroll
    for (int s = 0; s < 4; ++s) {
#pragma unroll
        for (int r = 0; r < 8; ++r) {
            int row = m0 + mgroup + s * 16 + r + hi * 8;
            float val = acc[s][r] + bval;
            if constexpr (OUT_BF16) {
                ((bf16_t*)Cout)[(size_t)row * N + col] = (bf16_t)val;
            } else {
                ((float*)Cout)[(size_t)row * N + col] = val;
            }
        }
    }
}

// ---------------------------------------------------------------------------
// RoPE in place on bf16 q/k. One thread per (token, head, pair).
// post_scale folds the 1/sqrt(HD) softmax scale into q.
// ---------------------------------------------------------------------------
__global__ void rope_bf16(bf16_t* __restrict__ qk, int n_heads, float post_scale) {
    int idx = blockIdx.x * blockDim.x + threadIdx.x;
    int total = NTOK * n_heads * (HD / 2);
    if (idx >= total) return;
    int i = idx & 31;                    // pair index, HD/2 = 32
    int h = (idx >> 5) % n_heads;
    int t = idx / (32 * n_heads);
    int pos = t & (SEQ - 1);
    // theta_i = 10000^(-2i/64) = exp(-(2i/64)*ln(10000))
    float freq = __expf(-((float)(2 * i) / (float)HD) * 9.210340371976184f);
    float ang = (float)pos * freq;
    float s = __sinf(ang);
    float c = __cosf(ang);
    bf16_t* p = qk + ((size_t)t * n_heads + h) * HD + 2 * i;
    float x1 = (float)p[0];
    float x2 = (float)p[1];
    p[0] = (bf16_t)((x1 * c - x2 * s) * post_scale);
    p[1] = (bf16_t)((x1 * s + x2 * c) * post_scale);
}

// ---------------------------------------------------------------------------
// Causal flash attention, one wave per (b, h, 16-query tile).
// Scores: D(16x16) = Q(16x32) x K^T(32x16), two d-chunks; Q/K/V fragments all
// come straight from global as contiguous 16/32-byte per-lane runs (V via the
// pre-transposed vT). Online softmax with shfl_xor reductions over 16-lane
// halves; P re-laid out C->A through a 1 KB LDS tile; 4 PV WMMAs per block.
// All key-block loads are provably in bounds (jb <= SEQ-32), no clamps.
// ---------------------------------------------------------------------------
__global__ __launch_bounds__(32) void flash_attn_bf16(
    const bf16_t* __restrict__ q,
    const bf16_t* __restrict__ k,
    const bf16_t* __restrict__ vT,
    bf16_t* __restrict__ o)
{
    __shared__ bf16_t Ps[16][40];

    const int lane = threadIdx.x & 31;
    const int lo   = lane & 15;
    const int hi   = lane >> 4;

    const int qtiles = SEQ / 16;
    const int qt  = blockIdx.x % qtiles;
    const int h   = (blockIdx.x / qtiles) % NH;
    const int b   = blockIdx.x / (qtiles * NH);
    const int kvh = h / (NH / NKV);
    const int q0  = qt * 16;

    // Q fragments (A layout): lane lo = query row, 32B contiguous run per d-chunk
    FragB fq0, fq1;
    {
        const bf16_t* qrow = q + (size_t)(b * SEQ + q0 + lo) * QDIM + h * HD;
        fq0.u[0] = *(const uint4*)(qrow + 0 * 32 + hi * 16);
        fq0.u[1] = *(const uint4*)(qrow + 0 * 32 + hi * 16 + 8);
        fq1.u[0] = *(const uint4*)(qrow + 1 * 32 + hi * 16);
        fq1.u[1] = *(const uint4*)(qrow + 1 * 32 + hi * 16 + 8);
    }

    v8f oacc[4] = {{}, {}, {}, {}};
    float mrow[8], lrow[8];
#pragma unroll
    for (int r = 0; r < 8; ++r) { mrow[r] = -1e30f; lrow[r] = 0.0f; }

    for (int jb = 0; jb < q0 + 16; jb += 32) {
        // K^T fragments (B layout): lane lo = key row, 32B contiguous run per chunk
        FragB fk00, fk01, fk10, fk11;
        {
            const bf16_t* kp0 = k + (size_t)(b * SEQ + jb + lo) * KVDIM + kvh * HD;
            const bf16_t* kp1 = kp0 + (size_t)16 * KVDIM;
            fk00.u[0] = *(const uint4*)(kp0 + 0 * 32 + hi * 16);
            fk00.u[1] = *(const uint4*)(kp0 + 0 * 32 + hi * 16 + 8);
            fk01.u[0] = *(const uint4*)(kp0 + 1 * 32 + hi * 16);
            fk01.u[1] = *(const uint4*)(kp0 + 1 * 32 + hi * 16 + 8);
            fk10.u[0] = *(const uint4*)(kp1 + 0 * 32 + hi * 16);
            fk10.u[1] = *(const uint4*)(kp1 + 0 * 32 + hi * 16 + 8);
            fk11.u[0] = *(const uint4*)(kp1 + 1 * 32 + hi * 16);
            fk11.u[1] = *(const uint4*)(kp1 + 1 * 32 + hi * 16 + 8);
        }

        v8f s0 = {}, s1 = {};
        s0 = wmma_bf16(fq0.v, fk00.v, s0);
        s0 = wmma_bf16(fq1.v, fk01.v, s0);
        s1 = wmma_bf16(fq0.v, fk10.v, s1);
        s1 = wmma_bf16(fq1.v, fk11.v, s1);

        // mask + online softmax (row M = r + 8*hi lives in 16 lanes of one half)
#pragma unroll
        for (int r = 0; r < 8; ++r) {
            int qrow = q0 + r + hi * 8;
            int key0 = jb + lo;
            int key1 = jb + 16 + lo;
            float a0 = (key0 <= qrow) ? s0[r] : -1e30f;   // scale pre-folded into q
            float a1 = (key1 <= qrow) ? s1[r] : -1e30f;
            float t = fmaxf(a0, a1);
#pragma unroll
            for (int off = 1; off < 16; off <<= 1) t = fmaxf(t, __shfl_xor(t, off, 32));
            float mnew  = fmaxf(mrow[r], t);
            float alpha = __expf(mrow[r] - mnew);
            float p0 = __expf(a0 - mnew);
            float p1 = __expf(a1 - mnew);
            float ps = p0 + p1;
#pragma unroll
            for (int off = 1; off < 16; off <<= 1) ps += __shfl_xor(ps, off, 32);
            lrow[r] = lrow[r] * alpha + ps;
            mrow[r] = mnew;
#pragma unroll
            for (int c = 0; c < 4; ++c) oacc[c][r] *= alpha;
            Ps[r + hi * 8][lo]      = (bf16_t)p0;
            Ps[r + hi * 8][16 + lo] = (bf16_t)p1;
        }
        __syncthreads();

        // P as A fragment (16 queries x 32 keys)
        FragB fp;
    fp.u[0] = *(const uint4*)&Ps[lo][hi * 16];
    fp.u[1] = *(const uint4*)&Ps[lo][hi * 16 + 8];

        // V as B fragments from vT: lane lo = d column, elements = keys
        // (contiguous 32B run starting at key jb + hi*16)
#pragma unroll
        for (int c = 0; c < 4; ++c) {
            FragB fv;
            const bf16_t* vrow =
                vT + (((size_t)b * NKV + kvh) * HD + c * 16 + lo) * SEQ + jb + hi * 16;
            fv.u[0] = *(const uint4*)(vrow);
            fv.u[1] = *(const uint4*)(vrow + 8);
            oacc[c] = wmma_bf16(fp.v, fv.v, oacc[c]);
        }
        __syncthreads();
    }

    // finalize: divide by row sums, store bf16 for the output projection
#pragma unroll
    for (int r = 0; r < 8; ++r) {
        float inv = 1.0f / lrow[r];
        int tok = b * SEQ + q0 + r + hi * 8;
#pragma unroll
        for (int c = 0; c < 4; ++c) {
            o[(size_t)tok * QDIM + h * HD + c * 16 + lo] = (bf16_t)(oacc[c][r] * inv);
        }
    }
}

// ---------------------------------------------------------------------------
// Host-side orchestration
// ---------------------------------------------------------------------------
extern "C" void kernel_launch(void* const* d_in, const int* in_sizes, int n_in,
                              void* d_out, int out_size, void* d_ws, size_t ws_size,
                              hipStream_t stream) {
    (void)in_sizes; (void)n_in; (void)out_size; (void)ws_size;

    const float* x  = (const float*)d_in[0];
    const float* Wq = (const float*)d_in[1];
    const float* Wk = (const float*)d_in[2];
    const float* Wv = (const float*)d_in[3];
    const float* bv = (const float*)d_in[4];
    const float* Wo = (const float*)d_in[5];
    const float* bo = (const float*)d_in[6];
    float* out = (float*)d_out;

    char* ws = (char*)d_ws;
    size_t off = 0;
    bf16_t* xb  = (bf16_t*)(ws + off); off += (size_t)NTOK * DIM * 2;
    bf16_t* Wqb = (bf16_t*)(ws + off); off += (size_t)DIM * QDIM * 2;
    bf16_t* Wkb = (bf16_t*)(ws + off); off += (size_t)DIM * KVDIM * 2;
    bf16_t* Wvb = (bf16_t*)(ws + off); off += (size_t)DIM * KVDIM * 2;
    bf16_t* Wob = (bf16_t*)(ws + off); off += (size_t)QDIM * DIM * 2;
    bf16_t* qb  = (bf16_t*)(ws + off); off += (size_t)NTOK * QDIM * 2;
    bf16_t* kb  = (bf16_t*)(ws + off); off += (size_t)NTOK * KVDIM * 2;
    bf16_t* vb  = (bf16_t*)(ws + off); off += (size_t)NTOK * KVDIM * 2;
    bf16_t* vTb = (bf16_t*)(ws + off); off += (size_t)NTOK * KVDIM * 2;
    bf16_t* ab  = (bf16_t*)(ws + off); off += (size_t)NTOK * QDIM * 2;

    const int T = 256;
    auto cdiv = [](int a, int b) { return (a + b - 1) / b; };

    cvt_f32_to_bf16<<<cdiv(NTOK * DIM, T), T, 0, stream>>>(x,  xb,  NTOK * DIM);
    cvt_f32_to_bf16<<<cdiv(DIM * QDIM, T), T, 0, stream>>>(Wq, Wqb, DIM * QDIM);
    cvt_f32_to_bf16<<<cdiv(DIM * KVDIM, T), T, 0, stream>>>(Wk, Wkb, DIM * KVDIM);
    cvt_f32_to_bf16<<<cdiv(DIM * KVDIM, T), T, 0, stream>>>(Wv, Wvb, DIM * KVDIM);
    cvt_f32_to_bf16<<<cdiv(QDIM * DIM, T), T, 0, stream>>>(Wo, Wob, QDIM * DIM);

    dim3 gq(QDIM / 64, NTOK / 128);    // (32, 32)
    dim3 gkv(KVDIM / 64, NTOK / 128);  // (8, 32)

    gemm_bf16_wmma<true,  false><<<gq,  256, 0, stream>>>(xb, Wqb, nullptr, qb, NTOK, QDIM,  DIM);
    gemm_bf16_wmma<true,  false><<<gkv, 256, 0, stream>>>(xb, Wkb, nullptr, kb, NTOK, KVDIM, DIM);
    gemm_bf16_wmma<true,  true ><<<gkv, 256, 0, stream>>>(xb, Wvb, bv,      vb, NTOK, KVDIM, DIM);

    rope_bf16<<<cdiv(NTOK * NH  * (HD / 2), T), T, 0, stream>>>(qb, NH,  0.125f);
    rope_bf16<<<cdiv(NTOK * NKV * (HD / 2), T), T, 0, stream>>>(kb, NKV, 1.0f);

    transpose_v<<<cdiv(NTOK * KVDIM, T), T, 0, stream>>>(vb, vTb);

    flash_attn_bf16<<<BATCH * NH * (SEQ / 16), 32, 0, stream>>>(qb, kb, vTb, ab);

    gemm_bf16_wmma<false, true ><<<gq, 256, 0, stream>>>(ab, Wob, bo, out, NTOK, QDIM, DIM);
}
